// DyadicTransform1D_7249904795701
// MI455X (gfx1250) — matrix operand order repaired and checked by reading, not compiled, not run on hardware
//
#include <hip/hip_runtime.h>

typedef __attribute__((ext_vector_type(16))) _Float16 v16h;
typedef __attribute__((ext_vector_type(8)))  _Float16 v8h;
typedef __attribute__((ext_vector_type(4)))  _Float16 v4h;
typedef __attribute__((ext_vector_type(8)))  float    v8f;
typedef __attribute__((ext_vector_type(4)))  float    v4f;

#define N_DIM 512
#define M_DIM 128
#define SCALE_INV (1.0f / 1048576.0f)   // 1 / 2^20
#define LDS_ROW 520                      // 512 + 8 halves pad: 1040B row stride
                                         // -> 260-dword advance -> 4 banks/row,
                                         // rows 0..15 land on distinct banks
#define XRE_OFF 0
#define XIM_OFF (M_DIM * N_DIM)          // halves

// ---------------------------------------------------------------------------
// Kernel 1: one-time f32 -> f16 conversion of X into workspace.
// 131072 elements, 8 per thread, 16384 threads.
// ---------------------------------------------------------------------------
__global__ __launch_bounds__(256)
void x_to_f16_kernel(const float* __restrict__ x_re,
                     const float* __restrict__ x_im,
                     _Float16* __restrict__ ws)
{
    const int t = blockIdx.x * blockDim.x + threadIdx.x;   // 0..16383
    const bool re = (t < 8192);
    const float* src = re ? x_re : x_im;
    const int base = (t & 8191) * 8;
    v4f a = *(const v4f*)(src + base);
    v4f b = *(const v4f*)(src + base + 4);
    v8h o;
#pragma unroll
    for (int i = 0; i < 4; ++i) {
        o[i]     = (_Float16)a[i];
        o[4 + i] = (_Float16)b[i];
    }
    *(v8h*)(ws + (re ? XRE_OFF : XIM_OFF) + base) = o;
}

// 32B LDS B-fragment as two 16B loads (16B-aligned with the 520-half row pad).
__device__ __forceinline__ v16h lds_load16(const _Float16* p) {
    v8h lo = *(const v8h*)(p);
    v8h hi = *(const v8h*)(p + 8);
    v16h r;
#pragma unroll
    for (int i = 0; i < 8; ++i) { r[i] = lo[i]; r[8 + i] = hi[i]; }
    return r;
}

// A-fragment: two 8-half runs (16B each) from the f16 X row.
__device__ __forceinline__ v16h a_frag(const _Float16* p) {
    v8h lo = *(const v8h*)(p);        // halves 0..7  : K = kb + h*16 ..
    v8h hi = *(const v8h*)(p + 8);    // halves 8..15 : K = kb + h*16 + 8 ..
    v16h r;
#pragma unroll
    for (int i = 0; i < 8; ++i) { r[i] = lo[i]; r[8 + i] = hi[i]; }
    return r;
}

// ---------------------------------------------------------------------------
// Kernel 2: coefficient build into LDS + WMMA GEMM.
//   out_re = Xre*A^T + Xim*B^T ; out_im = Xre*C^T + Xim*A^T
//   A = s(1+LG), B = sL(2+LG), C = sG  per (k,j) of the 512x512 tables.
// Grid (32 n-tiles, 2), 128 threads = 4 waves; wave w owns m-tile by*4+w.
// ---------------------------------------------------------------------------
__global__ __launch_bounds__(128)
void dyadic_wmma_kernel(const _Float16* __restrict__ x16,
                        const float* __restrict__ lambdas,
                        const float* __restrict__ gammas,
                        const unsigned char* __restrict__ negates,
                        float* __restrict__ out)
{
    __shared__ _Float16 ldsA[16 * LDS_ROW];
    __shared__ _Float16 ldsB[16 * LDS_ROW];
    __shared__ _Float16 ldsC[16 * LDS_ROW];

    const int tid = threadIdx.x;
    const int n0  = blockIdx.x * 16;

    // ---------------- Phase 1: build coefficient tiles (once per n-tile pair) --
    {
        const int row    = tid >> 3;         // 0..15: coefficient row n0+row
        const int kstart = (tid & 7) * 64;   // eighth of the 512-long row
        const float*         lamp = lambdas + (n0 + row) * N_DIM + kstart;
        const float*         gamp = gammas  + (n0 + row) * N_DIM + kstart;
        const unsigned char* negp = negates + (n0 + row) * N_DIM + kstart;
        _Float16* la = ldsA + row * LDS_ROW + kstart;
        _Float16* lb = ldsB + row * LDS_ROW + kstart;
        _Float16* lc = ldsC + row * LDS_ROW + kstart;

#pragma unroll 4
        for (int c = 0; c < 16; ++c) {
            v4f l4 = *(const v4f*)(lamp + 4 * c);
            v4f g4 = *(const v4f*)(gamp + 4 * c);
            unsigned int nb = *(const unsigned int*)(negp + 4 * c);
            v4h ah, bh, ch;
#pragma unroll
            for (int i = 0; i < 4; ++i) {
                float l  = l4[i] * SCALE_INV;
                float g  = g4[i] * SCALE_INV;
                float s  = ((nb >> (8 * i)) & 1u) ? -1.0f : 1.0f;
                float lg = l * g;
                ah[i] = (_Float16)(s * (1.0f + lg));
                bh[i] = (_Float16)(s * (l * (2.0f + lg)));
                ch[i] = (_Float16)(s * g);
            }
            *(v4h*)(la + 4 * c) = ah;
            *(v4h*)(lb + 4 * c) = bh;
            *(v4h*)(lc + 4 * c) = ch;
        }
    }
    __syncthreads();

    // ---------------- Phase 2: WMMA GEMM loop ---------------------------------
    const int wave = tid >> 5;            // 0..3
    const int lane = tid & 31;
    const int h    = lane >> 4;           // K half-group select
    const int ln   = lane & 15;
    const int m0   = (blockIdx.y * 4 + wave) * 16;

    // A-operand rows (f16, preconverted): lane holds X row m0+ln.
    const _Float16* xrp = x16 + XRE_OFF + (m0 + ln) * N_DIM + h * 16;
    const _Float16* xip = x16 + XIM_OFF + (m0 + ln) * N_DIM + h * 16;

    // B-operand: lane holds coeff column n0+ln; halves = K kb+h*16 .. +16.
    const _Float16* pA = ldsA + ln * LDS_ROW + h * 16;
    const _Float16* pB = ldsB + ln * LDS_ROW + h * 16;
    const _Float16* pC = ldsC + ln * LDS_ROW + h * 16;

    v8f acc_re = {};
    v8f acc_im = {};

    for (int kt = 0; kt < 16; ++kt) {
        const int kb = kt * 32;

        if (kt < 15) {  // next K-tile of X toward the WGP (global_prefetch_b8)
            __builtin_prefetch(xrp + kb + 32, 0, 1);
            __builtin_prefetch(xip + kb + 32, 0, 1);
        }

        v16h ar = a_frag(xrp + kb);
        v16h ai = a_frag(xip + kb);
        v16h bA = lds_load16(pA + kb);
        v16h bB = lds_load16(pB + kb);
        v16h bC = lds_load16(pC + kb);

        acc_re = __builtin_amdgcn_wmma_f32_16x16x32_f16(false, ar, false, bA,
                                                        (short)0, acc_re, false, false);
        acc_re = __builtin_amdgcn_wmma_f32_16x16x32_f16(false, ai, false, bB,
                                                        (short)0, acc_re, false, false);
        acc_im = __builtin_amdgcn_wmma_f32_16x16x32_f16(false, ar, false, bC,
                                                        (short)0, acc_im, false, false);
        acc_im = __builtin_amdgcn_wmma_f32_16x16x32_f16(false, ai, false, bA,
                                                        (short)0, acc_im, false, false);
    }

    // D-matrix 16x16 f32 layout: VGPR r -> M = r + 8*h, N = ln.
    const int ocol = n0 + ln;
    const int orow = m0 + h * 8;
#pragma unroll
    for (int r = 0; r < 8; ++r) {
        out[(orow + r) * N_DIM + ocol]                 = acc_re[r];
        out[M_DIM * N_DIM + (orow + r) * N_DIM + ocol] = acc_im[r];
    }
}

extern "C" void kernel_launch(void* const* d_in, const int* in_sizes, int n_in,
                              void* d_out, int out_size, void* d_ws, size_t ws_size,
                              hipStream_t stream) {
    (void)in_sizes; (void)n_in; (void)out_size; (void)ws_size;
    const float*         x_re = (const float*)d_in[0];
    const float*         x_im = (const float*)d_in[1];
    const float*         lam  = (const float*)d_in[2];
    const float*         gam  = (const float*)d_in[3];
    const unsigned char* neg  = (const unsigned char*)d_in[4];
    float*     out = (float*)d_out;
    _Float16*  x16 = (_Float16*)d_ws;    // 256 KB: f16 Xre | Xim

    // One-time X precision conversion (removes all cvt VALU from the GEMM loop).
    x_to_f16_kernel<<<64, 256, 0, stream>>>(x_re, x_im, x16);

    dim3 grid(N_DIM / 16, M_DIM / 64);   // 32 n-tiles x 2
    dyadic_wmma_kernel<<<grid, 128, 0, stream>>>(x16, lam, gam, neg, out);
}
